// EllipticalAttention_55611236549124
// MI455X (gfx1250) — matrix operand order repaired and checked by reading, not compiled, not run on hardware
//
#include <hip/hip_runtime.h>
#include <hip/hip_bf16.h>

// ---------------------------------------------------------------------------
// Elliptical attention for MI455X (gfx1250, wave32, WMMA 16x16x32 f16,
// async global->LDS double buffering).  B=2, H=8, S=2048, D=64.
// ---------------------------------------------------------------------------

typedef __attribute__((ext_vector_type(16))) _Float16 v16h;
typedef __attribute__((ext_vector_type(8)))  _Float16 v8h;
typedef __attribute__((ext_vector_type(8)))  float    v8f;
typedef __attribute__((ext_vector_type(4)))  int      v4i;

#define B_DIM 2
#define H_DIM 8
#define S_LEN 2048
#define D_DIM 64
#define JC    32      // j-chunk (K dim of the P@V WMMA)
#define ROWS  64      // query rows per workgroup (4 waves x 16-row strip)

__device__ __forceinline__ v8f wmma_f16(v16h a, v16h b, v8f c) {
  return __builtin_amdgcn_wmma_f32_16x16x32_f16(false, a, false, b, (short)0, c,
                                                false, false);
}

// ---- async global->LDS (ASYNCcnt) -----------------------------------------
__device__ __forceinline__ void async_b128(const float* g, float* l) {
#if __has_builtin(__builtin_amdgcn_global_load_async_to_lds_b128)
  __builtin_amdgcn_global_load_async_to_lds_b128(
      (__attribute__((address_space(1))) v4i*)(uintptr_t)g,
      (__attribute__((address_space(3))) v4i*)(uintptr_t)l, 0, 0);
#else
  asm volatile("global_load_async_to_lds_b128 %0, %1, off" ::
               "v"((unsigned)(uintptr_t)l),
               "v"((unsigned long long)(uintptr_t)g)
               : "memory");
#endif
}

__device__ __forceinline__ void wait_async0() {
#if __has_builtin(__builtin_amdgcn_s_wait_asynccnt)
  __builtin_amdgcn_s_wait_asynccnt(0);
#else
  asm volatile("s_wait_asynccnt 0" ::: "memory");
#endif
}

// issue one K/V chunk (JC x D floats each) into f32 staging buffers
__device__ __forceinline__ void issue_chunk(const float* Kg, const float* Vg,
                                            int j0, float* bK, float* bV,
                                            int tid) {
  const float* gk = Kg + (size_t)j0 * D_DIM;
  const float* gv = Vg + (size_t)j0 * D_DIM;
#pragma unroll
  for (int k = 0; k < (JC * D_DIM) / (128 * 4); ++k) {  // 4 x b128 per thread
    int off = (tid + k * 128) * 4;
    async_b128(gk + off, bK + off);
    async_b128(gv + off, bV + off);
  }
}

// ---- f16 fragment builders (two contiguous ds_load_b128 each) -------------
__device__ __forceinline__ v16h frag_combine(v8h lo, v8h hi) {
  v16h r;
#pragma unroll
  for (int i = 0; i < 8; ++i) { r[i] = lo[i]; r[i + 8] = hi[i]; }
  return r;
}

// A (16x32 f16) from row-major [M][K] tile, stride ld halves.
// Lane m=lane&15 holds row m; halves are 16 consecutive K values starting at 16g.
__device__ __forceinline__ v16h a_frag(const _Float16* lds, int row0, int ld,
                                       int kbase, int lane) {
  int m = lane & 15, g = (lane >> 4) & 1;
  const _Float16* p = lds + (row0 + m) * ld + kbase + 16 * g;
  v8h lo = *(const v8h*)p;
  v8h hi = *(const v8h*)(p + 8);
  return frag_combine(lo, hi);
}

// B (32x16 f16) from [N][K] tile, stride ld halves.  Lane n=lane&15 holds
// column n; halves are 16 consecutive K values starting at 16g.
__device__ __forceinline__ v16h b_frag(const _Float16* lds, int ld, int kbase,
                                       int nbase, int lane) {
  int n = lane & 15, g = (lane >> 4) & 1;
  const _Float16* p = lds + (nbase + n) * ld + kbase + 16 * g;
  v8h lo = *(const v8h*)p;
  v8h hi = *(const v8h*)(p + 8);
  return frag_combine(lo, hi);
}

__device__ __forceinline__ v16h ones_frag() {
  v16h o;
#pragma unroll
  for (int i = 0; i < 16; ++i) o[i] = (_Float16)1.0f;
  return o;
}

// ---------------------------------------------------------------------------
__global__ void zero_kernel(float* p, int n) {
  int i = blockIdx.x * blockDim.x + threadIdx.x;
  if (i < n) p[i] = 0.0f;
}

// ---------------------------------------------------------------------------
// Kernel 1: W sensitivity.  grid = (B*H*D, S/ROWS), block = 128 (4 waves).
// attn±[s,j] ∝ exp(scale*Q·K ± K[j,e]/128 - 1e-6*(1-mask[j])); accumulate
// H± = P± @ V and rowsum(P±) with WMMA; atomicAdd sum|H+-H-|/256.
// ---------------------------------------------------------------------------
__global__ __launch_bounds__(128) void wsens_kernel(
    const float* __restrict__ Q, const float* __restrict__ K,
    const float* __restrict__ V, const float* __restrict__ mask,
    float* __restrict__ Wraw) {
  const int bh = blockIdx.x >> 6;
  const int e  = blockIdx.x & 63;
  const int b  = bh >> 3;
  const int s0 = blockIdx.y * ROWS;
  const int tid  = threadIdx.x;
  const int wave = tid >> 5;
  const int lane = tid & 31;

  __shared__ __align__(16) float    stgK[2][JC * D_DIM];   // f32 staging (async dst)
  __shared__ __align__(16) float    stgV[2][JC * D_DIM];
  __shared__ __align__(32) _Float16 sQh[ROWS * D_DIM];     // [m][d]
  __shared__ __align__(32) _Float16 sKh[JC * D_DIM];       // [j][d]
  __shared__ __align__(32) _Float16 sVt[D_DIM * JC];       // [d][j] (transposed)
  __shared__ __align__(32) _Float16 sPp[4][16 * JC];       // per-wave P+ [m][j]
  __shared__ __align__(32) _Float16 sPm[4][16 * JC];       // per-wave P- [m][j]

  const float scale = 0.125f;
  const float* Qg = Q + ((size_t)bh * S_LEN + s0) * D_DIM;
  const float* Kg = K + (size_t)bh * S_LEN * D_DIM;
  const float* Vg = V + (size_t)bh * S_LEN * D_DIM;

  issue_chunk(Kg, Vg, 0, stgK[0], stgV[0], tid);           // prologue DMA

  for (int idx = tid; idx < ROWS * D_DIM; idx += 128)
    sQh[idx] = (_Float16)(Qg[idx] * scale);
  __syncthreads();

  const v16h aQ0 = a_frag(sQh, wave * 16, D_DIM, 0, lane);
  const v16h aQ1 = a_frag(sQh, wave * 16, D_DIM, 32, lane);
  const v16h bOne = ones_frag();

  const v8f zacc = {0.f, 0.f, 0.f, 0.f, 0.f, 0.f, 0.f, 0.f};
  v8f accP[4], accM[4], dnP = zacc, dnM = zacc;
#pragma unroll
  for (int t = 0; t < 4; ++t) { accP[t] = zacc; accM[t] = zacc; }

  const int n = lane & 15;
  const int g = (lane >> 4) & 1;

  int cur = 0;
  for (int j0 = 0; j0 < S_LEN; j0 += JC, cur ^= 1) {
    wait_async0();                 // this thread's chunk `cur` has landed
    __syncthreads();               // ...for every thread; prev compute done
    if (j0 + JC < S_LEN)
      issue_chunk(Kg, Vg, j0 + JC, stgK[cur ^ 1], stgV[cur ^ 1], tid);

    // convert K chunk to f16 [j][d]
    for (int idx = tid * 2; idx < JC * D_DIM; idx += 256) {
      sKh[idx]     = (_Float16)stgK[cur][idx];
      sKh[idx + 1] = (_Float16)stgK[cur][idx + 1];
    }
    // transpose V chunk to f16 [d][j]
    for (int idx = tid; idx < JC * D_DIM; idx += 128) {
      int j = idx >> 6, d = idx & 63;
      sVt[d * JC + j] = (_Float16)stgV[cur][idx];
    }
    __syncthreads();

#pragma unroll
    for (int t2 = 0; t2 < 2; ++t2) {
      v16h bK0 = b_frag(sKh, D_DIM, 0, t2 * 16, lane);
      v16h bK1 = b_frag(sKh, D_DIM, 32, t2 * 16, lane);
      v8f cs = zacc;
      cs = wmma_f16(aQ0, bK0, cs);
      cs = wmma_f16(aQ1, bK1, cs);
      const int jc = t2 * 16 + n;
      const float delta = stgK[cur][jc * D_DIM + e] * (1.0f / 128.0f);
      const float pen = 1e-6f * (1.0f - mask[b * S_LEN + j0 + jc]);
#pragma unroll
      for (int r = 0; r < 8; ++r) {
        float s = cs[r] - pen;
        sPp[wave][(r + 8 * g) * JC + jc] = (_Float16)__expf(s + delta);
        sPm[wave][(r + 8 * g) * JC + jc] = (_Float16)__expf(s - delta);
      }
    }
    asm volatile("s_wait_dscnt 0" ::: "memory");  // wave-local LDS RAW

    v16h aPp = a_frag(sPp[wave], 0, JC, 0, lane);
    v16h aPm = a_frag(sPm[wave], 0, JC, 0, lane);
    dnP = wmma_f16(aPp, bOne, dnP);
    dnM = wmma_f16(aPm, bOne, dnM);
#pragma unroll
    for (int t = 0; t < 4; ++t) {
      v16h bV = b_frag(sVt, JC, 0, t * 16, lane);
      accP[t] = wmma_f16(aPp, bV, accP[t]);
      accM[t] = wmma_f16(aPm, bV, accM[t]);
    }
  }

  float local = 0.0f;
#pragma unroll
  for (int r = 0; r < 8; ++r) {
    float ip = 1.0f / dnP[r];
    float im = 1.0f / dnM[r];
#pragma unroll
    for (int t = 0; t < 4; ++t)
      local += fabsf(accP[t][r] * ip - accM[t][r] * im);
  }
#pragma unroll
  for (int off = 16; off > 0; off >>= 1) local += __shfl_xor(local, off, 32);
  if (lane == 0)
    atomicAdd(&Wraw[bh * D_DIM + e], local * (1.0f / 256.0f));  // 1/(2*t*S)
}

// ---------------------------------------------------------------------------
// Kernel 2: W normalization.  grid = 16, block = 64.
// ---------------------------------------------------------------------------
__global__ void wfin_kernel(const float* __restrict__ Wraw,
                            float* __restrict__ Wfin) {
  __shared__ float red[64];
  const int bh = blockIdx.x;
  const int e = threadIdx.x;
  float w = Wraw[bh * D_DIM + e];
  red[e] = w;
  __syncthreads();
#pragma unroll
  for (int off = 32; off > 0; off >>= 1) {
    if (e < off) red[e] = fmaxf(red[e], red[e + off]);
    __syncthreads();
  }
  Wfin[bh * D_DIM + e] = w / red[0];
}

// ---------------------------------------------------------------------------
// Kernel 3: weighted attention forward.  grid = (B*H, S/ROWS), block = 128.
// ---------------------------------------------------------------------------
__global__ __launch_bounds__(128) void fwd_kernel(
    const float* __restrict__ Q, const float* __restrict__ K,
    const float* __restrict__ V, const float* __restrict__ mask,
    const float* __restrict__ W, float* __restrict__ X) {
  const int bh = blockIdx.x;
  const int b  = bh >> 3;
  const int s0 = blockIdx.y * ROWS;
  const int tid  = threadIdx.x;
  const int wave = tid >> 5;
  const int lane = tid & 31;

  __shared__ __align__(16) float    stgK[2][JC * D_DIM];
  __shared__ __align__(16) float    stgV[2][JC * D_DIM];
  __shared__ __align__(32) _Float16 sQh[ROWS * D_DIM];
  __shared__ __align__(32) _Float16 sKh[JC * D_DIM];
  __shared__ __align__(32) _Float16 sVt[D_DIM * JC];
  __shared__ __align__(32) _Float16 sP[4][16 * JC];

  const float scale = 0.125f;
  const float* Qg = Q + ((size_t)bh * S_LEN + s0) * D_DIM;
  const float* Kg = K + (size_t)bh * S_LEN * D_DIM;
  const float* Vg = V + (size_t)bh * S_LEN * D_DIM;

  issue_chunk(Kg, Vg, 0, stgK[0], stgV[0], tid);

  for (int idx = tid; idx < ROWS * D_DIM; idx += 128) {
    int d = idx & (D_DIM - 1);
    sQh[idx] = (_Float16)(Qg[idx] * W[bh * D_DIM + d] * scale);
  }
  __syncthreads();

  const v16h aQ0 = a_frag(sQh, wave * 16, D_DIM, 0, lane);
  const v16h aQ1 = a_frag(sQh, wave * 16, D_DIM, 32, lane);
  const v16h bOne = ones_frag();

  const v8f zacc = {0.f, 0.f, 0.f, 0.f, 0.f, 0.f, 0.f, 0.f};
  v8f acc[4], dn = zacc;
#pragma unroll
  for (int t = 0; t < 4; ++t) acc[t] = zacc;

  const int n = lane & 15;
  const int g = (lane >> 4) & 1;

  int cur = 0;
  for (int j0 = 0; j0 < S_LEN; j0 += JC, cur ^= 1) {
    wait_async0();
    __syncthreads();
    if (j0 + JC < S_LEN)
      issue_chunk(Kg, Vg, j0 + JC, stgK[cur ^ 1], stgV[cur ^ 1], tid);

    for (int idx = tid * 2; idx < JC * D_DIM; idx += 256) {
      sKh[idx]     = (_Float16)stgK[cur][idx];
      sKh[idx + 1] = (_Float16)stgK[cur][idx + 1];
    }
    for (int idx = tid; idx < JC * D_DIM; idx += 128) {
      int j = idx >> 6, d = idx & 63;
      sVt[d * JC + j] = (_Float16)stgV[cur][idx];
    }
    __syncthreads();

#pragma unroll
    for (int t2 = 0; t2 < 2; ++t2) {
      v16h bK0 = b_frag(sKh, D_DIM, 0, t2 * 16, lane);
      v16h bK1 = b_frag(sKh, D_DIM, 32, t2 * 16, lane);
      v8f cs = zacc;
      cs = wmma_f16(aQ0, bK0, cs);
      cs = wmma_f16(aQ1, bK1, cs);
      const int jc = t2 * 16 + n;
      const float pen = 1e6f * (1.0f - mask[b * S_LEN + j0 + jc]);
#pragma unroll
      for (int r = 0; r < 8; ++r)
        sP[wave][(r + 8 * g) * JC + jc] = (_Float16)__expf(cs[r] - pen);
    }
    asm volatile("s_wait_dscnt 0" ::: "memory");

    v16h aP = a_frag(sP[wave], 0, JC, 0, lane);
    dn = wmma_f16(aP, bOne, dn);
#pragma unroll
    for (int t = 0; t < 4; ++t) {
      v16h bV = b_frag(sVt, JC, 0, t * 16, lane);
      acc[t] = wmma_f16(aP, bV, acc[t]);
    }
  }

  float* Xg = X + ((size_t)bh * S_LEN + s0 + wave * 16) * D_DIM;
#pragma unroll
  for (int r = 0; r < 8; ++r) {
    const int m = r + 8 * g;
    const float inv = 1.0f / dn[r];
#pragma unroll
    for (int t = 0; t < 4; ++t)
      Xg[m * D_DIM + t * 16 + n] = acc[t][r] * inv;
  }
}

// ---------------------------------------------------------------------------
extern "C" void kernel_launch(void* const* d_in, const int* in_sizes, int n_in,
                              void* d_out, int out_size, void* d_ws,
                              size_t ws_size, hipStream_t stream) {
  (void)in_sizes; (void)n_in; (void)out_size; (void)ws_size;
  const float* Q    = (const float*)d_in[0];
  const float* K    = (const float*)d_in[1];
  const float* V    = (const float*)d_in[2];
  const float* mask = (const float*)d_in[3];
  float* X = (float*)d_out;

  float* Wraw = (float*)d_ws;                       // [B*H*D] = 1024 floats
  float* Wfin = Wraw + B_DIM * H_DIM * D_DIM;       // [B*H*D]
  const int wn = B_DIM * H_DIM * D_DIM;

  zero_kernel<<<(wn + 255) / 256, 256, 0, stream>>>(Wraw, wn);

  dim3 gW(B_DIM * H_DIM * D_DIM, S_LEN / ROWS);     // (1024, 32)
  wsens_kernel<<<gW, 128, 0, stream>>>(Q, K, V, mask, Wraw);

  wfin_kernel<<<B_DIM * H_DIM, D_DIM, 0, stream>>>(Wraw, Wfin);

  dim3 gF(B_DIM * H_DIM, S_LEN / ROWS);             // (16, 32)
  fwd_kernel<<<gF, 128, 0, stream>>>(Q, K, V, mask, Wfin, X);
}
